// MyGRU_73323681677972
// MI455X (gfx1250) — matrix operand order repaired and checked by reading, not compiled, not run on hardware
//
#include <hip/hip_runtime.h>

// ---------------------------------------------------------------------------
// GRU on MI455X (gfx1250), wave32 + WMMA bf16.
//
// Decomposition: batch (64) split into 4 persistent workgroups of 16 rows
// (WMMA M tile). The 2048-step recurrence has NO inter-workgroup dependency;
// H[16][512] lives in LDS (single buffer), weights stream from L2 in a
// pre-swizzled bf16 layout matching the WMMA B-operand register order, so
// every weight fetch is a coalesced dwordx4 load straight into matrix regs.
// X projections are precomputed by a WMMA GEMM kernel and stored pre-swizzled
// in the C-operand order (accumulator init == two coalesced b128 loads).
// Per-lane old-H values are carried in registers across time steps, so the
// gate/blend math touches no LDS and H needs no double buffer.
// ---------------------------------------------------------------------------

typedef __bf16 bf16_t;
typedef __attribute__((ext_vector_type(16))) __bf16 v16bf;
typedef __attribute__((ext_vector_type(8)))  __bf16 v8bf;
typedef __attribute__((ext_vector_type(8)))  float  v8f;

#define BATCH   64
#define SEQLEN  2048
#define INSZ    256
#define HID     512
#define OUTSZ   256
#define BT      16          // batch tile = WMMA M
#define NBT     4           // 64/16
#define NJ      32          // N tiles over hidden (512/16)
#define NKH     16          // K tiles over hidden (512/32)
#define NKI     8           // K tiles over input  (256/32)
#define NJO     16          // N tiles over output (256/16)
#define LPITCH  520         // padded LDS row pitch (bf16) for K=512 (bank-skewed)
#define XPITCH  264         // padded LDS row pitch (bf16) for K=256

static __device__ __forceinline__ v8f wmma_bf16(v16bf a, v16bf b, v8f c) {
  // D = A(16x32 bf16) * B(32x16 bf16) + C(16x16 f32)
  return __builtin_amdgcn_wmma_f32_16x16x32_bf16(false, a, false, b,
                                                 (short)0, c, false, false);
}

// A-operand (16-bit 16x32) from an LDS row, per ISA layout:
// lanes 0-15: M=lane, K = {k0..k0+7, k0+16..k0+23}; lanes 16-31: +8 on K.
// Caller passes row = &LDS[lane&15][0], kbase = kk*32 + ((lane&16)?8:0).
static __device__ __forceinline__ v16bf load_a_tile(const bf16_t* __restrict__ row,
                                                    int kbase) {
  v8bf lo = *(const v8bf*)(row + kbase);        // ds_load_b128
  v8bf hi = *(const v8bf*)(row + kbase + 16);   // ds_load_b128
  v16bf a;
#pragma unroll
  for (int e = 0; e < 8; ++e) { a[e] = lo[e]; a[e + 8] = hi[e]; }
  return a;
}

static __device__ __forceinline__ float fast_sigmoid(float x) {
  return 1.0f / (1.0f + __expf(-x));
}

// ---------------------------------------------------------------------------
// Pack a fp32 weight matrix W[K][N] into the WMMA B-operand register order:
// flat = ((j*nkk + kk)*32 + lane)*16 + i ; n = j*16 + (lane&15),
// k = kk*32 + ((lane&16)?8:0) + (i<8 ? i : i+8)   (B: same K order as A tile).
// ---------------------------------------------------------------------------
__global__ void pack_b_kernel(const float* __restrict__ W, bf16_t* __restrict__ dst,
                              int nelem, int nkk, int Ntot) {
  int idx = blockIdx.x * 256 + threadIdx.x;
  if (idx >= nelem) return;
  int i    = idx & 15;
  int lane = (idx >> 4) & 31;
  int jk   = idx >> 9;
  int kk   = jk % nkk;
  int j    = jk / nkk;
  int n = j * 16 + (lane & 15);
  int k = kk * 32 + ((lane & 16) ? 8 : 0) + (i < 8 ? i : i + 8);
  dst[idx] = (bf16_t)W[(size_t)k * Ntot + n];
}

// Concatenate gate biases into one buffer so the GEMM kernel has a single
// global base pointer (avoids pointer-select -> FLAT loads in the hot loop).
__global__ void pack_bias_kernel(const float* __restrict__ bz,
                                 const float* __restrict__ br,
                                 const float* __restrict__ bh,
                                 float* __restrict__ bcat) {
  int i = blockIdx.x * 256 + threadIdx.x;
  if (i < HID)            bcat[i] = bz[i];
  else if (i < 2 * HID)   bcat[i] = br[i - HID];
  else if (i < 3 * HID)   bcat[i] = bh[i - 2 * HID];
}

// ---------------------------------------------------------------------------
// X projections: X[g] = inputs @ Wx[g] + b[g], stored pre-swizzled in
// C-operand order: block(t,bt,g,j) of 512 bf16 laid out [lane][v].
// One block per (t, batch-tile); 8 waves x 12 (g,j) tiles.
// The full A operand (16x256) is hoisted into 64 VGPRs once per wave.
// ---------------------------------------------------------------------------
__global__ __launch_bounds__(256)
void xproj_kernel(const float* __restrict__ inputs, const bf16_t* __restrict__ wxpack,
                  const float* __restrict__ bcat, bf16_t* __restrict__ xbuf) {
  int t  = blockIdx.x;
  int bt = blockIdx.y;
  __shared__ bf16_t lx[BT][XPITCH];

  int tid = threadIdx.x;
  { // stage inputs[bt*16+row][t][0..255] -> LDS (fp32 -> bf16)
    int row = tid >> 4;
    int kc  = (tid & 15) * 16;
    const float* src = inputs + ((size_t)(bt * BT + row) * SEQLEN + t) * INSZ + kc;
#pragma unroll
    for (int c = 0; c < 16; ++c) lx[row][kc + c] = (bf16_t)src[c];
  }
  __syncthreads();

  int lane = tid & 31, w = tid >> 5;
  int n15 = lane & 15;
  int kb  = (lane & 16) ? 8 : 0;
  const v16bf* wv = (const v16bf*)wxpack;

  // hoist all K-tiles of A into registers (8 tiles x 8 VGPRs)
  v16bf atile[NKI];
#pragma unroll
  for (int kk = 0; kk < NKI; ++kk)
    atile[kk] = load_a_tile(&lx[n15][0], kk * 32 + kb);

#pragma unroll
  for (int pi = 0; pi < 12; ++pi) {
    int p = w * 12 + pi;           // 0..95
    int g = p >> 5;                // 0..2
    int j = p & 31;                // 0..31
    float bias = bcat[g * HID + j * 16 + n15];   // single-base global load
    v8f acc;
#pragma unroll
    for (int e = 0; e < 8; ++e) acc[e] = bias;
#pragma unroll
    for (int kk = 0; kk < NKI; ++kk) {
      v16bf b = wv[(((size_t)(g * NJ + j)) * NKI + kk) * 32 + lane];
      acc = wmma_bf16(atile[kk], b, acc);
    }
    bf16_t* blk = xbuf + ((((size_t)t * NBT + bt) * 3 + g) * NJ + j) * 512;
    v8bf o;
#pragma unroll
    for (int e = 0; e < 8; ++e) o[e] = (bf16_t)acc[e];
    *(v8bf*)(blk + lane * 8) = o;
  }
}

// ---------------------------------------------------------------------------
// Persistent GRU scan: 4 workgroups (one per 16-row batch tile), 8 waves.
// Per step: fused Z/R GEMM (K-outer, 1 A load -> 8 WMMAs), sigmoid,
// RH = R*H -> LDS bf16, barrier, H~ GEMM + tanh + blend, barrier.
// Old H at this lane's C-layout positions is carried in hprev[] registers.
// ---------------------------------------------------------------------------
__global__ __launch_bounds__(256)
void gru_scan_kernel(const bf16_t* __restrict__ whpack,
                     const bf16_t* __restrict__ xbuf,
                     bf16_t* __restrict__ hfin) {
  __shared__ bf16_t Hls[BT][LPITCH];
  __shared__ bf16_t RH[BT][LPITCH];

  int tid  = threadIdx.x;
  int lane = tid & 31, w = tid >> 5;
  int bt   = blockIdx.x;
  int n15  = lane & 15;
  int kb   = (lane & 16) ? 8 : 0;
  int mrow = (lane & 16) ? 8 : 0;   // C-operand: M = mrow + v

  for (int idx = tid; idx < BT * LPITCH; idx += 256)
    (&Hls[0][0])[idx] = (bf16_t)0.0f;
  __syncthreads();

  const v16bf* wv = (const v16bf*)whpack;

  float hprev[4][8];                // H_old at this lane's (jj,v) positions
#pragma unroll
  for (int jj = 0; jj < 4; ++jj)
#pragma unroll
    for (int v = 0; v < 8; ++v) hprev[jj][v] = 0.0f;

  for (int t = 0; t < SEQLEN; ++t) {
    const bf16_t* xbase = xbuf + ((size_t)t * NBT + bt) * 3 * NJ * 512;

    // prefetch next step's X blocks for this wave (global_prefetch_b8)
    if (t + 1 < SEQLEN) {
      const bf16_t* xn = xbuf + ((size_t)(t + 1) * NBT + bt) * 3 * NJ * 512;
#pragma unroll
      for (int g = 0; g < 3; ++g)
#pragma unroll
        for (int jj = 0; jj < 4; ++jj)
          __builtin_prefetch(xn + ((size_t)(g * NJ + (w * 4 + jj))) * 512 + lane * 16, 0, 1);
    }

    // ---- Phase 1: Z and R GEMMs (C init = precomputed xz/xr + bias) ----
    v8f accZ[4], accR[4];
#pragma unroll
    for (int jj = 0; jj < 4; ++jj) {
      int j = w * 4 + jj;
      v8bf cz = *((const v8bf*)(xbase + ((size_t)(0 * NJ + j)) * 512) + lane);
      v8bf cr = *((const v8bf*)(xbase + ((size_t)(1 * NJ + j)) * 512) + lane);
#pragma unroll
      for (int e = 0; e < 8; ++e) { accZ[jj][e] = (float)cz[e]; accR[jj][e] = (float)cr[e]; }
    }
#pragma unroll 8
    for (int kk = 0; kk < NKH; ++kk) {
      v16bf a = load_a_tile(&Hls[n15][0], kk * 32 + kb);
#pragma unroll
      for (int jj = 0; jj < 4; ++jj) {
        int j = w * 4 + jj;
        v16bf bz = wv[(((size_t)(0 * NJ + j)) * NKH + kk) * 32 + lane];
        accZ[jj] = wmma_bf16(a, bz, accZ[jj]);
        v16bf br = wv[(((size_t)(1 * NJ + j)) * NKH + kk) * 32 + lane];
        accR[jj] = wmma_bf16(a, br, accR[jj]);
      }
    }
    // gates; write RH = R * H_old (H_old from registers); keep Z in accZ
#pragma unroll
    for (int jj = 0; jj < 4; ++jj) {
      int j = w * 4 + jj;
#pragma unroll
      for (int v = 0; v < 8; ++v) {
        int m = mrow + v;
        int n = j * 16 + n15;
        float z = fast_sigmoid(accZ[jj][v]);
        float r = fast_sigmoid(accR[jj][v]);
        accZ[jj][v] = z;
        RH[m][n] = (bf16_t)(r * hprev[jj][v]);
      }
    }
    __syncthreads();   // RH complete; phase-1 A reads of Hls complete

    // ---- Phase 2: H~ = tanh(xh + (R*H) @ Whh); blend; write new H ----
    v8f accH[4];
#pragma unroll
    for (int jj = 0; jj < 4; ++jj) {
      int j = w * 4 + jj;
      v8bf ch = *((const v8bf*)(xbase + ((size_t)(2 * NJ + j)) * 512) + lane);
#pragma unroll
      for (int e = 0; e < 8; ++e) accH[jj][e] = (float)ch[e];
    }
#pragma unroll 8
    for (int kk = 0; kk < NKH; ++kk) {
      v16bf a = load_a_tile(&RH[n15][0], kk * 32 + kb);
#pragma unroll
      for (int jj = 0; jj < 4; ++jj) {
        int j = w * 4 + jj;
        v16bf bh = wv[(((size_t)(2 * NJ + j)) * NKH + kk) * 32 + lane];
        accH[jj] = wmma_bf16(a, bh, accH[jj]);
      }
    }
#pragma unroll
    for (int jj = 0; jj < 4; ++jj) {
      int j = w * 4 + jj;
#pragma unroll
      for (int v = 0; v < 8; ++v) {
        int m = mrow + v;
        int n = j * 16 + n15;
        float ht = tanhf(accH[jj][v]);
        float z  = accZ[jj][v];
        float hn = z * hprev[jj][v] + (1.0f - z) * ht;
        hprev[jj][v] = hn;
        Hls[m][n] = (bf16_t)hn;
      }
    }
    __syncthreads();   // new H visible before next step's GEMMs
  }

  // Hls holds the final hidden state
  for (int idx = tid; idx < BT * HID; idx += 256) {
    int m = idx >> 9, k = idx & 511;
    hfin[(size_t)(bt * BT + m) * HID + k] = Hls[m][k];
  }
}

// ---------------------------------------------------------------------------
// out = H_final @ Whq + bq   (64x512 @ 512x256, WMMA, single WG)
// ---------------------------------------------------------------------------
__global__ __launch_bounds__(256)
void final_proj_kernel(const bf16_t* __restrict__ hfin,
                       const bf16_t* __restrict__ whqpack,
                       const float* __restrict__ bq, float* __restrict__ out) {
  __shared__ bf16_t lh[BATCH][LPITCH];
  int tid = threadIdx.x, lane = tid & 31, w = tid >> 5;

  for (int idx = tid; idx < BATCH * (HID / 8); idx += 256) {
    int m  = idx / (HID / 8);
    int kc = (idx % (HID / 8)) * 8;
    *(v8bf*)&lh[m][kc] = *(const v8bf*)&hfin[(size_t)m * HID + kc];
  }
  __syncthreads();

  int n15 = lane & 15;
  int kb  = (lane & 16) ? 8 : 0;
  int mrow = (lane & 16) ? 8 : 0;
  const v16bf* wv = (const v16bf*)whqpack;

#pragma unroll
  for (int ti = 0; ti < 8; ++ti) {
    int tt = w * 8 + ti;     // 0..63
    int mt = tt >> 4;        // 0..3  (batch tile)
    int jt = tt & 15;        // 0..15 (output N tile)
    float bias = bq[jt * 16 + n15];
    v8f acc;
#pragma unroll
    for (int e = 0; e < 8; ++e) acc[e] = bias;
#pragma unroll 8
    for (int kk = 0; kk < NKH; ++kk) {
      v16bf a = load_a_tile(&lh[mt * 16 + n15][0], kk * 32 + kb);
      v16bf b = wv[(((size_t)jt) * NKH + kk) * 32 + lane];
      acc = wmma_bf16(a, b, acc);
    }
#pragma unroll
    for (int v = 0; v < 8; ++v) {
      int row = mt * 16 + mrow + v;
      out[(size_t)row * OUTSZ + jt * 16 + n15] = acc[v];
    }
  }
}

// ---------------------------------------------------------------------------
// Workspace layout (bytes):
//   whpack  @ 0          : 3*32*16*512*2 = 1,572,864
//   wxpack  @ 1,572,864  : 3*32* 8*512*2 =   786,432
//   whqpack @ 2,359,296  : 16*16*512*2   =   262,144
//   hfin    @ 2,621,440  : 64*512*2      =    65,536
//   bcat    @ 2,686,976  : 3*512*4       =     6,144
//   xbuf    @ 2,693,120  : 2048*4*3*32*512*2 = 805,306,368  (~771 MB total)
// ---------------------------------------------------------------------------
extern "C" void kernel_launch(void* const* d_in, const int* in_sizes, int n_in,
                              void* d_out, int out_size, void* d_ws, size_t ws_size,
                              hipStream_t stream) {
  (void)in_sizes; (void)n_in; (void)out_size; (void)ws_size;
  const float* inputs = (const float*)d_in[0];
  const float* Wxr = (const float*)d_in[1];
  const float* Whr = (const float*)d_in[2];
  const float* Wxz = (const float*)d_in[3];
  const float* Whz = (const float*)d_in[4];
  const float* Wxh = (const float*)d_in[5];
  const float* Whh = (const float*)d_in[6];
  const float* br  = (const float*)d_in[7];
  const float* bz  = (const float*)d_in[8];
  const float* bh  = (const float*)d_in[9];
  const float* Whq = (const float*)d_in[10];
  const float* bq  = (const float*)d_in[11];
  float* out = (float*)d_out;

  char* ws = (char*)d_ws;
  bf16_t* whpack  = (bf16_t*)(ws + 0);
  bf16_t* wxpack  = (bf16_t*)(ws + 1572864);
  bf16_t* whqpack = (bf16_t*)(ws + 2359296);
  bf16_t* hfin    = (bf16_t*)(ws + 2621440);
  float*  bcat    = (float*) (ws + 2686976);
  bf16_t* xbuf    = (bf16_t*)(ws + 2693120);

  const int WH_ELEMS = NJ * NKH * 512;   // 262144 per hidden matrix
  const int WX_ELEMS = NJ * NKI * 512;   // 131072 per input matrix
  const int WQ_ELEMS = NJO * NKH * 512;  // 131072

  // gate order g: 0=z, 1=r, 2=h
  pack_b_kernel<<<WH_ELEMS / 256, 256, 0, stream>>>(Whz, whpack + 0 * WH_ELEMS, WH_ELEMS, NKH, HID);
  pack_b_kernel<<<WH_ELEMS / 256, 256, 0, stream>>>(Whr, whpack + 1 * WH_ELEMS, WH_ELEMS, NKH, HID);
  pack_b_kernel<<<WH_ELEMS / 256, 256, 0, stream>>>(Whh, whpack + 2 * WH_ELEMS, WH_ELEMS, NKH, HID);
  pack_b_kernel<<<WX_ELEMS / 256, 256, 0, stream>>>(Wxz, wxpack + 0 * WX_ELEMS, WX_ELEMS, NKI, HID);
  pack_b_kernel<<<WX_ELEMS / 256, 256, 0, stream>>>(Wxr, wxpack + 1 * WX_ELEMS, WX_ELEMS, NKI, HID);
  pack_b_kernel<<<WX_ELEMS / 256, 256, 0, stream>>>(Wxh, wxpack + 2 * WX_ELEMS, WX_ELEMS, NKI, HID);
  pack_b_kernel<<<WQ_ELEMS / 256, 256, 0, stream>>>(Whq, whqpack,               WQ_ELEMS, NKH, OUTSZ);
  pack_bias_kernel<<<6, 256, 0, stream>>>(bz, br, bh, bcat);

  xproj_kernel<<<dim3(SEQLEN, NBT), 256, 0, stream>>>(inputs, wxpack, bcat, xbuf);
  gru_scan_kernel<<<NBT, 256, 0, stream>>>(whpack, xbuf, hfin);
  final_proj_kernel<<<1, 256, 0, stream>>>(hfin, whqpack, bq, out);
}
